// CIN_52725018526223
// MI455X (gfx1250) — matrix operand order repaired and checked by reading, not compiled, not run on hardware
//
#include <hip/hip_runtime.h>

// CDNA5 / gfx1250: wave32, WMMA bf16 16x16x32, f32 accumulate.
typedef __attribute__((ext_vector_type(16))) __bf16 v16bf;
typedef __attribute__((ext_vector_type(8)))  float  v8f;

#define BV    2048
#define NKB0  48          // ceil(1521/32)
#define NKB1  78          // 2496/32
#define K0    1521        // 39*39
#define K1    2496        // 39*64
#define OFF0  0
#define OFF1  (8 * NKB0 * 512)            // 196608 bf16 elems
#define OFF2  (OFF1 + 8 * NKB1 * 512)     // 516096 bf16 elems
// total pack: 835584 bf16 = 1,671,168 bytes in d_ws

__device__ __forceinline__ unsigned short f32_to_bf16_bits(float x) {
  unsigned int u = __float_as_uint(x);
  unsigned int lsb = (u >> 16) & 1u;
  u += 0x7fffu + lsb;                     // round-to-nearest-even
  return (unsigned short)(u >> 16);
}

// Pre-swizzle W (row-major K x 128, f32) into per-lane WMMA A-fragment order:
// pack[((t*nkb + kb)*32 + lane)*16 + e], lane fragment = 16 contiguous bf16.
// A 16x32 bf16 layout (matches B fill): lanes 0-15 hold K {0..15}, lanes 16-31
// hold K {16..31}; element e -> K = 16*half + e.
__global__ void cin_pack_w(const float* __restrict__ W,
                           unsigned short* __restrict__ pack,
                           int K, int nkb) {
  int idx = blockIdx.x * blockDim.x + threadIdx.x;
  int total = 8 * nkb * 512;
  if (idx >= total) return;
  int e    = idx & 15;
  int l    = (idx >> 4) & 31;
  int tkb  = idx >> 9;
  int kb   = tkb % nkb;
  int t    = tkb / nkb;
  int half = l >> 4;
  int v    = e >> 1;
  int koff = half * 16 + e; (void)v;
  int k    = kb * 32 + koff;
  int s    = t * 16 + (l & 15);
  float val = (k < K) ? W[k * 128 + s] : 0.0f;
  pack[idx] = f32_to_bf16_bits(val);
}

// One wave computes the full 128x16 layer output for one batch element.
// U: x[b] rows (f major, 16 cols, rows >= 39 zeroed), V: h rows, both LDS.
template <int FK_, int NKB_>
__device__ __forceinline__ void cin_layer(const unsigned short* __restrict__ packW,
                                          const float* __restrict__ U,
                                          const float* __restrict__ V,
                                          const float* __restrict__ bias,
                                          float* __restrict__ O,
                                          int lane) {
  const int half = lane >> 4;
  const int n    = lane & 15;                 // this lane's d column (B/C/D N index)
  const v16bf* __restrict__ Apack = (const v16bf*)packW;

  v8f zero = {0.f, 0.f, 0.f, 0.f, 0.f, 0.f, 0.f, 0.f};
  v8f acc[8];
#pragma unroll
  for (int t = 0; t < 8; ++t) acc[t] = zero;

  for (int kb = 0; kb < NKB_; ++kb) {
    // B-fragment: lane holds column n, K-run [k0, k0+16). (f,g) is uniform per
    // lane-half; consecutive k -> consecutive g.
    const int k0 = kb * 32 + half * 16;
    v16bf bfrag;
    if constexpr ((FK_ & 15) == 0) {
      // k0 is a multiple of 16 and 16 | FK_  =>  the 16-run never crosses an
      // f boundary: single u scalar, straight-line g indexing, no selects.
      const int f = k0 / FK_;
      const int g = k0 - f * FK_;
      const float u0 = U[f * 16 + n];
#pragma unroll
      for (int e = 0; e < 16; ++e) {
        float p = u0 * V[(g + e) * 16 + n];
        bfrag[e] = (__bf16)p;
      }
    } else {
      // General path (layer 0, FK_=39): at most one f wrap per 16-run.
      // K tail (k >= 1521) lands in zero-padded U rows -> product is 0,
      // no explicit predication needed.
      int f = k0 / FK_;
      int g = k0 - f * FK_;
      float u0 = U[f * 16 + n];
      float u1 = U[(f + 1) * 16 + n];          // safe: U padded+zeroed rows
      float uc = u0;
      int   gg = g;
#pragma unroll
      for (int e = 0; e < 16; ++e) {
        float p = uc * V[gg * 16 + n];
        bfrag[e] = (__bf16)p;
        ++gg;
        if (gg == FK_) { gg = 0; uc = u1; }
      }
    }
    // 8 WMMAs share this B fragment (A changes per s-tile).
#pragma unroll
    for (int t = 0; t < 8; ++t) {
      v16bf afrag = Apack[(t * NKB_ + kb) * 32 + lane];
      acc[t] = __builtin_amdgcn_wmma_f32_16x16x32_bf16(
          false, afrag, false, bfrag, (short)0, acc[t], false, false);
    }
  }

  // C/D layout: VGPR r -> M = r + 8*half, N = n. s = 16*t + M.
#pragma unroll
  for (int t = 0; t < 8; ++t) {
#pragma unroll
    for (int r = 0; r < 8; ++r) {
      int s = t * 16 + half * 8 + r;
      float val = acc[t][r] + bias[s];
      O[s * 16 + n] = fmaxf(val, 0.0f);
    }
  }
}

__device__ __forceinline__ float rowsum16(const float* __restrict__ p) {
  float s = 0.f;
#pragma unroll
  for (int i = 0; i < 16; ++i) s += p[i];
  return s;
}

__global__ void __launch_bounds__(128, 1) cin_main(
    const float* __restrict__ x,
    const unsigned short* __restrict__ pack,
    const float* __restrict__ b0,
    const float* __restrict__ b1,
    const float* __restrict__ b2,
    float* __restrict__ out) {
  __shared__ float sBias[128];
  __shared__ float sU[4][42 * 16];     // x[b]: 39 rows + zeroed pad rows 39..41
  __shared__ float sV[4][64 * 16];     // h for layers 1,2
  __shared__ float sO[4][128 * 16];    // per-layer output staging

  const int tid  = threadIdx.x;
  const int wid  = tid >> 5;
  const int lane = tid & 31;
  const int b    = blockIdx.x * 4 + wid;

  float* U = sU[wid];
  float* V = sV[wid];
  float* O = sO[wid];

  // Stage x[b]; zero the pad rows so the layer-0 K tail self-masks.
  for (int i = lane; i < 42 * 16; i += 32)
    U[i] = (i < 39 * 16) ? x[b * (39 * 16) + i] : 0.0f;
  if (tid < 128) sBias[tid] = b0[tid];
  __syncthreads();

  // ---- layer 0: h = x (V source == U), K = 39*39 ----
  cin_layer<39, NKB0>(pack + OFF0, U, U, sBias, O, lane);
  __syncthreads();

  for (int i = lane; i < 64 * 16; i += 32) V[i] = O[i];                  // h1
  for (int rr = lane; rr < 64; rr += 32)
    out[b * 256 + rr] = rowsum16(O + (64 + rr) * 16);                    // direct0
  if (tid < 128) sBias[tid] = b1[tid];
  __syncthreads();

  // ---- layer 1: K = 39*64 ----
  cin_layer<64, NKB1>(pack + OFF1, U, V, sBias, O, lane);
  __syncthreads();

  for (int i = lane; i < 64 * 16; i += 32) V[i] = O[i];                  // h2
  for (int rr = lane; rr < 64; rr += 32)
    out[b * 256 + 64 + rr] = rowsum16(O + (64 + rr) * 16);               // direct1
  if (tid < 128) sBias[tid] = b2[tid];
  __syncthreads();

  // ---- layer 2: K = 39*64, all 128 rows are direct ----
  cin_layer<64, NKB1>(pack + OFF2, U, V, sBias, O, lane);
  __syncthreads();

  for (int rr = lane; rr < 128; rr += 32)
    out[b * 256 + 128 + rr] = rowsum16(O + rr * 16);                     // direct2
}

extern "C" void kernel_launch(void* const* d_in, const int* in_sizes, int n_in,
                              void* d_out, int out_size, void* d_ws, size_t ws_size,
                              hipStream_t stream) {
  (void)in_sizes; (void)n_in; (void)out_size; (void)ws_size;
  const float* x  = (const float*)d_in[0];
  const float* W0 = (const float*)d_in[1];
  const float* W1 = (const float*)d_in[2];
  const float* W2 = (const float*)d_in[3];
  const float* b0 = (const float*)d_in[4];
  const float* b1 = (const float*)d_in[5];
  const float* b2 = (const float*)d_in[6];
  unsigned short* pack = (unsigned short*)d_ws;   // needs 1,671,168 B of d_ws

  cin_pack_w<<<(8 * NKB0 * 512 + 255) / 256, 256, 0, stream>>>(W0, pack + OFF0, K0, NKB0);
  cin_pack_w<<<(8 * NKB1 * 512 + 255) / 256, 256, 0, stream>>>(W1, pack + OFF1, K1, NKB1);
  cin_pack_w<<<(8 * NKB1 * 512 + 255) / 256, 256, 0, stream>>>(W2, pack + OFF2, K1, NKB1);
  cin_main<<<BV / 4, 128, 0, stream>>>(x, pack, b0, b1, b2, (float*)d_out);
}
